// Objects_netv2_85736137163369
// MI455X (gfx1250) — matrix operand order repaired and checked by reading, not compiled, not run on hardware
//
#include <hip/hip_runtime.h>
#include <hip/hip_bf16.h>
#include <math.h>

// ---------------------------------------------------------------------------
// Types for CDNA5 WMMA (wave32, 16x16x32 bf16 -> f32)
// ---------------------------------------------------------------------------
typedef __attribute__((ext_vector_type(16))) __bf16 v16bf;
typedef __attribute__((ext_vector_type(8)))  float  v8f;

#define TM  128
#define TN  128
#define TK  32
#define LDP 40   // padded LDS row stride (elements) -> conflict-free b128 reads

template <typename T> struct NCHUNK;
template <> struct NCHUNK<float>  { static constexpr int n = 4; };
template <> struct NCHUNK<__bf16> { static constexpr int n = 2; };

// ---------------------------------------------------------------------------
// Pipelined tile staging, full tiles only (no guards in the hot loop).
// 256 threads stage a 128row x 32k tile.
//   f32 source : 4 chunks/thread of float4 (4 values along k)
//   bf16 source: 2 chunks/thread of 16B   (8 values along k)
// ---------------------------------------------------------------------------
__device__ __forceinline__ void g2r(float4 (&r)[4], const float* g, int ldg,
                                    int k0, int tid) {
#pragma unroll
  for (int i = 0; i < 4; ++i) {
    int c = tid + i * 256, row = c >> 3, kc = (c & 7) << 2;
    r[i] = *(const float4*)(g + (size_t)row * ldg + k0 + kc);
  }
}
__device__ __forceinline__ void r2l(const float4 (&r)[4], __bf16* lds, int tid) {
#pragma unroll
  for (int i = 0; i < 4; ++i) {
    int c = tid + i * 256, row = c >> 3, kc = (c & 7) << 2;
    union { __bf16 h[4]; unsigned long long u; } pk;
    pk.h[0] = (__bf16)r[i].x; pk.h[1] = (__bf16)r[i].y;
    pk.h[2] = (__bf16)r[i].z; pk.h[3] = (__bf16)r[i].w;
    *(unsigned long long*)(lds + row * LDP + kc) = pk.u;   // ds_store_b64
  }
}
__device__ __forceinline__ void g2r(float4 (&r)[2], const __bf16* g, int ldg,
                                    int k0, int tid) {
#pragma unroll
  for (int i = 0; i < 2; ++i) {
    int c = tid + i * 256, row = c >> 2, kc = (c & 3) << 3;
    r[i] = *(const float4*)(g + (size_t)row * ldg + k0 + kc);
  }
}
__device__ __forceinline__ void r2l(const float4 (&r)[2], __bf16* lds, int tid) {
#pragma unroll
  for (int i = 0; i < 2; ++i) {
    int c = tid + i * 256, row = c >> 2, kc = (c & 3) << 3;
    *(float4*)(lds + row * LDP + kc) = r[i];               // ds_store_b128
  }
}

// Guarded tail stage (runs at most once per kernel, zero-fills k >= K).
__device__ __forceinline__ void stage_tail(__bf16* lds, const float* g, int ldg,
                                           int k0, int K, int tid) {
#pragma unroll
  for (int i = 0; i < 16; ++i) {
    int idx = tid + i * 256, row = idx >> 5, kc = idx & 31, k = k0 + kc;
    lds[row * LDP + kc] = (k < K) ? (__bf16)g[(size_t)row * ldg + k] : (__bf16)0.0f;
  }
}
__device__ __forceinline__ void stage_tail(__bf16* lds, const __bf16* g, int ldg,
                                           int k0, int K, int tid) {
#pragma unroll
  for (int i = 0; i < 16; ++i) {
    int idx = tid + i * 256, row = idx >> 5, kc = idx & 31, k = k0 + kc;
    lds[row * LDP + kc] = (k < K) ? g[(size_t)row * ldg + k] : (__bf16)0.0f;
  }
}

// ---------------------------------------------------------------------------
// Fragment loads from LDS per ISA 7.12.2 16-bit layouts.
// A 16x32: lanes 0-15 (M=lane): K{0..7,16..23}; lanes 16-31: K{8..15,24..31}
// B 32x16: lanes 0-15 (N=lane): K{0..7,16..23}; lanes 16-31: K{8..15,24..31}
// ---------------------------------------------------------------------------
__device__ __forceinline__ v16bf frag_a(const __bf16* tile, int lane) {
  int r = lane & 15, kq = (lane >> 4) << 3;                // 0 or 8
  const __bf16* p = tile + r * LDP + kq;
  union { v16bf v; struct { float4 lo, hi; } q; } u;
  u.q.lo = *(const float4*)p;                              // K: kq..kq+7
  u.q.hi = *(const float4*)(p + 16);                       // K: kq+16..kq+23
  return u.v;
}
__device__ __forceinline__ v16bf frag_b(const __bf16* tile, int lane) {
  int r = lane & 15, kq = (lane >> 4) << 3;                // 0 or 8
  const __bf16* p = tile + r * LDP + kq;
  union { v16bf v; struct { float4 lo, hi; } q; } u;
  u.q.lo = *(const float4*)p;                              // K: kq..kq+7
  u.q.hi = *(const float4*)(p + 16);                       // K: kq+16..kq+23
  return u.v;
}

// One K-step of the wave's 32x64 subtile: 2 A frags x 4 B frags = 8 WMMA.
__device__ __forceinline__ void mma_step(const __bf16* As, const __bf16* Bs,
                                         v8f (&acc)[2][4], int lane, int wm, int wn) {
  v16bf a0 = frag_a(As + (wm * 32 + 0)  * LDP, lane);
  v16bf a1 = frag_a(As + (wm * 32 + 16) * LDP, lane);
  v16bf b0 = frag_b(Bs + (wn * 64 + 0)  * LDP, lane);
  v16bf b1 = frag_b(Bs + (wn * 64 + 16) * LDP, lane);
  v16bf b2 = frag_b(Bs + (wn * 64 + 32) * LDP, lane);
  v16bf b3 = frag_b(Bs + (wn * 64 + 48) * LDP, lane);
  acc[0][0] = __builtin_amdgcn_wmma_f32_16x16x32_bf16(false, a0, false, b0, (short)0, acc[0][0], false, false);
  acc[0][1] = __builtin_amdgcn_wmma_f32_16x16x32_bf16(false, a0, false, b1, (short)0, acc[0][1], false, false);
  acc[0][2] = __builtin_amdgcn_wmma_f32_16x16x32_bf16(false, a0, false, b2, (short)0, acc[0][2], false, false);
  acc[0][3] = __builtin_amdgcn_wmma_f32_16x16x32_bf16(false, a0, false, b3, (short)0, acc[0][3], false, false);
  acc[1][0] = __builtin_amdgcn_wmma_f32_16x16x32_bf16(false, a1, false, b0, (short)0, acc[1][0], false, false);
  acc[1][1] = __builtin_amdgcn_wmma_f32_16x16x32_bf16(false, a1, false, b1, (short)0, acc[1][1], false, false);
  acc[1][2] = __builtin_amdgcn_wmma_f32_16x16x32_bf16(false, a1, false, b2, (short)0, acc[1][2], false, false);
  acc[1][3] = __builtin_amdgcn_wmma_f32_16x16x32_bf16(false, a1, false, b3, (short)0, acc[1][3], false, false);
}

// ---------------------------------------------------------------------------
// Templated GEMM:  C[M,N] = A[M,K] @ W[N,K]^T  (W pre-converted to bf16)
// EPI 0: store f32 raw
// EPI 1: y = relu(acc*scale[n]+shift[n]) -> store bf16 (feeds next GEMM)
// EPI 2: same epilogue -> store f32 (final output; bias folded into shift)
// 256 threads = 8 waves; wave grid 4x2; each wave 32x64 (2x4 WMMA tiles).
// Double-buffered LDS: global loads for tile kt+1 fly over the WMMAs of kt.
// ---------------------------------------------------------------------------
template <typename TA, int EPI>
__global__ __launch_bounds__(256)
void gemm_wmma(const TA* __restrict__ A, const __bf16* __restrict__ W,
               void* __restrict__ Cout,
               const float* __restrict__ scale, const float* __restrict__ shift,
               int M, int N, int K) {
  __shared__ __bf16 As[2][TM * LDP];
  __shared__ __bf16 Bs[2][TN * LDP];

  const int tid   = threadIdx.x;
  const int lane  = tid & 31;
  const int wave  = tid >> 5;
  const int wm    = wave >> 1;                 // 0..3
  const int wn    = wave & 1;                  // 0..1
  const int tileM = blockIdx.y * TM;
  const int tileN = blockIdx.x * TN;

  const TA*     Ab = A + (size_t)tileM * K;
  const __bf16* Wb = W + (size_t)tileN * K;

  v8f acc[2][4];
#pragma unroll
  for (int i = 0; i < 2; ++i)
#pragma unroll
    for (int j = 0; j < 4; ++j) acc[i][j] = (v8f){0, 0, 0, 0, 0, 0, 0, 0};

  constexpr int NA = NCHUNK<TA>::n;
  float4 ra[NA], rb[2];

  const int nk    = (K + TK - 1) / TK;
  const int nfull = K / TK;                    // tiles 0..nfull-1 need no guards

  // prologue: stage tile 0
  if (nfull > 0) {
    g2r(ra, Ab, K, 0, tid); g2r(rb, Wb, K, 0, tid);
    r2l(ra, As[0], tid);    r2l(rb, Bs[0], tid);
  } else {
    stage_tail(As[0], Ab, K, 0, K, tid);
    stage_tail(Bs[0], Wb, K, 0, K, tid);
  }
  __syncthreads();

  // main pipeline over full tiles
  int kt = 0;
  for (; kt + 1 < nfull; ++kt) {
    const int nxt = (kt + 1) * TK;
    g2r(ra, Ab, K, nxt, tid);                  // next tile loads in flight
    g2r(rb, Wb, K, nxt, tid);
    mma_step(As[kt & 1], Bs[kt & 1], acc, lane, wm, wn);
    r2l(ra, As[(kt + 1) & 1], tid);            // convert + ds_store other buffer
    r2l(rb, Bs[(kt + 1) & 1], tid);
    __syncthreads();
  }
  // optional tail tile (only fc1, K=688) — guarded path runs exactly once
  if (nk > nfull && nfull > 0) {
    mma_step(As[kt & 1], Bs[kt & 1], acc, lane, wm, wn);
    stage_tail(As[(kt + 1) & 1], Ab, K, nfull * TK, K, tid);
    stage_tail(Bs[(kt + 1) & 1], Wb, K, nfull * TK, K, tid);
    __syncthreads();
    ++kt;
  }
  // last tile
  mma_step(As[kt & 1], Bs[kt & 1], acc, lane, wm, wn);

  // C/D layout: lanes 0-15 vgpr v -> (M=v, N=lane); lanes 16-31 -> (M=8+v, N=lane-16)
  const int nl = lane & 15;
  const int mh = (lane >> 4) << 3;
#pragma unroll
  for (int i = 0; i < 2; ++i) {
    const int mbase = tileM + wm * 32 + i * 16 + mh;
#pragma unroll
    for (int j = 0; j < 4; ++j) {
      const int n = tileN + wn * 64 + j * 16 + nl;
      float s = 1.0f, t = 0.0f;
      if (EPI != 0) { s = scale[n]; t = shift[n]; }
#pragma unroll
      for (int v = 0; v < 8; ++v) {
        float y = acc[i][j][v];
        if (EPI != 0) y = fmaxf(fmaf(y, s, t), 0.0f);
        const size_t off = (size_t)(mbase + v) * N + n;
        if (EPI == 1) ((__bf16*)Cout)[off] = (__bf16)y;
        else          ((float*)Cout)[off]  = y;
      }
    }
  }
}

// ---------------------------------------------------------------------------
// Prep kernels
// ---------------------------------------------------------------------------
__global__ void cvt_bf16_kernel(const float* __restrict__ src,
                                __bf16* __restrict__ dst, int n) {
  int i = blockIdx.x * 256 + threadIdx.x;
  if (i < n) dst[i] = (__bf16)src[i];
}

// scale = g*rsqrt(v+eps); shift = (bias - mean)*scale + b   (bias optional)
__global__ void bn_prep_kernel(const float* __restrict__ g, const float* __restrict__ b,
                               const float* __restrict__ m, const float* __restrict__ v,
                               const float* __restrict__ bias,
                               float* __restrict__ scale, float* __restrict__ shift, int n) {
  int i = blockIdx.x * 256 + threadIdx.x;
  if (i < n) {
    float s  = g[i] * rsqrtf(v[i] + 1e-5f);
    float bi = bias ? bias[i] : 0.0f;
    scale[i] = s;
    shift[i] = (bi - m[i]) * s + b[i];
  }
}

// ---------------------------------------------------------------------------
// Fused attention stage: one block per batch row b (1024 blocks, 256 threads)
// Builds x[10][688] in LDS, channel attention, spatial attention, sum over N.
// ---------------------------------------------------------------------------
#define NS 10
#define FD 688

__global__ __launch_bounds__(256)
void attention_kernel(const float* __restrict__ obj, const float* __restrict__ bbox,
                      const float* __restrict__ we,  const float* __restrict__ sen,
                      const float* __restrict__ bod,
                      const float* __restrict__ Wc1, const float* __restrict__ Wc2,
                      const float* __restrict__ Wsa,
                      float* __restrict__ xsum) {
  __shared__ float x[NS][FD];
  __shared__ float red[256];
  __shared__ float avgv[NS], maxv[NS], attc[NS];
  __shared__ float srow[2][FD];

  const int b   = blockIdx.x;
  const int tid = threadIdx.x;

  // gather the concatenated feature vector
  for (int idx = tid; idx < NS * FD; idx += 256) {
    int n = idx / FD, f = idx - n * FD;
    int row = b * NS + n;
    float v;
    if      (f < 128) v = obj[(size_t)row * 128 + f];
    else if (f < 132) v = bbox[(size_t)row * 4 + (f - 128)];
    else if (f < 432) v = we[(size_t)row * 300 + (f - 132)];
    else if (f < 560) v = sen[(size_t)row * 128 + (f - 432)];
    else              v = bod[(size_t)b * 128 + (f - 560)];
    x[n][f] = v;
  }
  __syncthreads();

  // per-slice mean and max over features
  for (int n = 0; n < NS; ++n) {
    float s = 0.0f, m = -INFINITY;
    for (int f = tid; f < FD; f += 256) { float v = x[n][f]; s += v; m = fmaxf(m, v); }
    red[tid] = s; __syncthreads();
    for (int off = 128; off > 0; off >>= 1) { if (tid < off) red[tid] += red[tid + off]; __syncthreads(); }
    if (tid == 0) avgv[n] = red[0] * (1.0f / FD);
    __syncthreads();
    red[tid] = m; __syncthreads();
    for (int off = 128; off > 0; off >>= 1) { if (tid < off) red[tid] = fmaxf(red[tid], red[tid + off]); __syncthreads(); }
    if (tid == 0) maxv[n] = red[0];
    __syncthreads();
  }

  // tiny shared MLP 10->5->10 on both branches, sigmoid of sum
  if (tid == 0) {
    float h[5];
#pragma unroll
    for (int j = 0; j < 5; ++j) {
      float s1 = 0.0f, s2 = 0.0f;
#pragma unroll
      for (int i = 0; i < NS; ++i) { s1 += avgv[i] * Wc1[j * NS + i]; s2 += maxv[i] * Wc1[j * NS + i]; }
      h[j] = fmaxf(s1, 0.0f) + fmaxf(s2, 0.0f);   // relu each branch, W2 is linear
    }
#pragma unroll
    for (int i = 0; i < NS; ++i) {
      float s = 0.0f;
#pragma unroll
      for (int j = 0; j < 5; ++j) s += h[j] * Wc2[i * 5 + j];
      attc[i] = 1.0f / (1.0f + __expf(-s));
    }
  }
  __syncthreads();

  // apply channel attention
  for (int idx = tid; idx < NS * FD; idx += 256) {
    int n = idx / FD;
    x[n][idx - n * FD] *= attc[n];
  }
  __syncthreads();

  // per-feature mean/max over the N axis
  for (int f = tid; f < FD; f += 256) {
    float s = 0.0f, m = -INFINITY;
#pragma unroll
    for (int n = 0; n < NS; ++n) { float v = x[n][f]; s += v; m = fmaxf(m, v); }
    srow[0][f] = s * (1.0f / NS);
    srow[1][f] = m;
  }
  __syncthreads();

  // 7-tap conv (pad 3), sigmoid, apply + sum over N
  for (int f = tid; f < FD; f += 256) {
    float a = 0.0f;
#pragma unroll
    for (int j = 0; j < 7; ++j) {
      int ff = f + j - 3;
      if (ff >= 0 && ff < FD) a += srow[0][ff] * Wsa[j] + srow[1][ff] * Wsa[7 + j];
    }
    float sig = 1.0f / (1.0f + __expf(-a));
    float s = 0.0f;
#pragma unroll
    for (int n = 0; n < NS; ++n) s += x[n][f];
    xsum[(size_t)b * FD + f] = s * sig;
  }
}

// ---------------------------------------------------------------------------
// Host-side launch
// ---------------------------------------------------------------------------
extern "C" void kernel_launch(void* const* d_in, const int* in_sizes, int n_in,
                              void* d_out, int out_size, void* d_ws, size_t ws_size,
                              hipStream_t stream) {
  const float* feat = (const float*)d_in[0];   // [1024,10,1024]
  const float* bbox = (const float*)d_in[1];   // [1024,10,4]
  const float* wemb = (const float*)d_in[2];   // [1024,10,300]
  const float* sent = (const float*)d_in[3];   // [1024,10,4096]
  const float* body = (const float*)d_in[4];   // [1024,2048]
  const float* W1o = (const float*)d_in[5];
  const float *g_o = (const float*)d_in[6], *b_o = (const float*)d_in[7],
              *m_o = (const float*)d_in[8], *v_o = (const float*)d_in[9];
  const float* W2o = (const float*)d_in[10];
  const float* W1s = (const float*)d_in[11];
  const float *g_s = (const float*)d_in[12], *b_s = (const float*)d_in[13],
              *m_s = (const float*)d_in[14], *v_s = (const float*)d_in[15];
  const float* W2s = (const float*)d_in[16];
  const float* W1b = (const float*)d_in[17];
  const float *g_b = (const float*)d_in[18], *b_b = (const float*)d_in[19],
              *m_b = (const float*)d_in[20], *v_b = (const float*)d_in[21];
  const float* W2b = (const float*)d_in[22];
  const float* Wc1 = (const float*)d_in[23];
  const float* Wc2 = (const float*)d_in[24];
  const float* Wsa = (const float*)d_in[25];
  const float* Wf  = (const float*)d_in[26];
  const float* bf  = (const float*)d_in[27];
  const float *g_f = (const float*)d_in[28], *b_f = (const float*)d_in[29],
              *m_f = (const float*)d_in[30], *v_f = (const float*)d_in[31];

  const int BN10 = 10240;  // B*N rows
  const int B    = 1024;

  // workspace carve-out (256B aligned)
  uintptr_t p = (uintptr_t)d_ws;
  auto alloc = [&](size_t bytes) -> void* {
    uintptr_t r = (p + 255) & ~(uintptr_t)255;
    p = r + bytes;
    return (void*)r;
  };
  __bf16* w1s = (__bf16*)alloc((size_t)512 * 4096 * 2);
  __bf16* w2s = (__bf16*)alloc((size_t)128 * 512  * 2);
  __bf16* w1o = (__bf16*)alloc((size_t)128 * 1024 * 2);
  __bf16* w2o = (__bf16*)alloc((size_t)128 * 128  * 2);
  __bf16* w1b = (__bf16*)alloc((size_t)256 * 2048 * 2);
  __bf16* w2b = (__bf16*)alloc((size_t)128 * 256  * 2);
  __bf16* wfb = (__bf16*)alloc((size_t)256 * 688  * 2);
  float* sc_s = (float*)alloc(512 * 4); float* sh_s = (float*)alloc(512 * 4);
  float* sc_o = (float*)alloc(128 * 4); float* sh_o = (float*)alloc(128 * 4);
  float* sc_b = (float*)alloc(256 * 4); float* sh_b = (float*)alloc(256 * 4);
  float* sc_f = (float*)alloc(256 * 4); float* sh_f = (float*)alloc(256 * 4);
  __bf16* Hs  = (__bf16*)alloc((size_t)BN10 * 512 * 2);
  __bf16* Ho  = (__bf16*)alloc((size_t)BN10 * 128 * 2);
  __bf16* Hb  = (__bf16*)alloc((size_t)B    * 256 * 2);
  float*  Ssen = (float*)alloc((size_t)BN10 * 128 * 4);
  float*  Sobj = (float*)alloc((size_t)BN10 * 128 * 4);
  float*  Sbod = (float*)alloc((size_t)B    * 128 * 4);
  float*  Xsum = (float*)alloc((size_t)B    * 688 * 4);
  (void)ws_size; (void)n_in; (void)in_sizes; (void)out_size;

  auto cvt = [&](const float* s, __bf16* d, int n) {
    cvt_bf16_kernel<<<(n + 255) / 256, 256, 0, stream>>>(s, d, n);
  };
  cvt(W1s, w1s, 512 * 4096); cvt(W2s, w2s, 128 * 512);
  cvt(W1o, w1o, 128 * 1024); cvt(W2o, w2o, 128 * 128);
  cvt(W1b, w1b, 256 * 2048); cvt(W2b, w2b, 128 * 256);
  cvt(Wf,  wfb, 256 * 688);

  bn_prep_kernel<<<2, 256, 0, stream>>>(g_s, b_s, m_s, v_s, nullptr, sc_s, sh_s, 512);
  bn_prep_kernel<<<1, 256, 0, stream>>>(g_o, b_o, m_o, v_o, nullptr, sc_o, sh_o, 128);
  bn_prep_kernel<<<1, 256, 0, stream>>>(g_b, b_b, m_b, v_b, nullptr, sc_b, sh_b, 256);
  bn_prep_kernel<<<1, 256, 0, stream>>>(g_f, b_f, m_f, v_f, bf,      sc_f, sh_f, 256);

  // sentiment: 10240x4096 @ 4096x512 -> bn+relu -> bf16 ; then @ 512x128 -> f32
  gemm_wmma<float, 1><<<dim3(512 / TN, BN10 / TM), 256, 0, stream>>>(
      sent, w1s, Hs, sc_s, sh_s, BN10, 512, 4096);
  gemm_wmma<__bf16, 0><<<dim3(1, BN10 / TM), 256, 0, stream>>>(
      Hs, w2s, Ssen, nullptr, nullptr, BN10, 128, 512);

  // object: 10240x1024 @ 1024x128 -> bf16 ; then @ 128x128 -> f32
  gemm_wmma<float, 1><<<dim3(1, BN10 / TM), 256, 0, stream>>>(
      feat, w1o, Ho, sc_o, sh_o, BN10, 128, 1024);
  gemm_wmma<__bf16, 0><<<dim3(1, BN10 / TM), 256, 0, stream>>>(
      Ho, w2o, Sobj, nullptr, nullptr, BN10, 128, 128);

  // body: 1024x2048 @ 2048x256 -> bf16 ; then @ 256x128 -> f32
  gemm_wmma<float, 1><<<dim3(256 / TN, B / TM), 256, 0, stream>>>(
      body, w1b, Hb, sc_b, sh_b, B, 256, 2048);
  gemm_wmma<__bf16, 0><<<dim3(1, B / TM), 256, 0, stream>>>(
      Hb, w2b, Sbod, nullptr, nullptr, B, 128, 256);

  // fused attention (channel + spatial) and sum over N -> Xsum[1024,688]
  attention_kernel<<<B, 256, 0, stream>>>(Sobj, bbox, wemb, Ssen, Sbod,
                                          Wc1, Wc2, Wsa, Xsum);

  // fc1: 1024x688 @ 688x256, bias folded into shift, bn+relu -> f32 output
  gemm_wmma<float, 2><<<dim3(256 / TN, B / TM), 256, 0, stream>>>(
      Xsum, wfb, d_out, sc_f, sh_f, B, 256, 688);
}
